// LowRankAttention_12403865551225
// MI455X (gfx1250) — matrix operand order repaired and checked by reading, not compiled, not run on hardware
//
#include <hip/hip_runtime.h>

// Problem constants (fixed by the reference)
#define B_    2
#define N_    2048
#define D_    1024
#define RANK_ 256
#define HEADS_ 16
#define HS_   16
#define DH_   64
#define MTOT  (B_ * N_)          // 4096 rows
#define SCALE_ 0.25f             // 1/sqrt(HS)

typedef float  v8f  __attribute__((ext_vector_type(8)));
typedef float  v2f  __attribute__((ext_vector_type(2)));
typedef __bf16 v16bf __attribute__((ext_vector_type(16)));

union BF16x16 { v16bf v; unsigned short u[16]; };

__device__ __forceinline__ unsigned short f32_to_bf16(float f) {
  unsigned int u = __float_as_uint(f);
  u += 0x7FFFu + ((u >> 16) & 1u);     // round-to-nearest-even
  return (unsigned short)(u >> 16);
}

// 16-bit A-matrix 16x32 K-map: VGPR0-3 => K = e + 8*half, VGPR4-7 => K = e + 8 + 8*half
__device__ __forceinline__ int kmapA16(int e, int half) {
  return (e < 8) ? (e + 8 * half) : (e + 8 + 8 * half);
}

// ---------------------------------------------------------------------------
// Kernel 1: C = X * W^T   (X:[4096,1024] f32, W:[Nout,1024] f32)
// mode 0: write f32 row-major (qk projection)
// mode 1: write bf16 into [b][h][n][64] head layout (v projection)
// Block: 256 thr = 8 waves; block tile 64x64; wave tile 16x32; K step 32.
// ---------------------------------------------------------------------------
__global__ __launch_bounds__(256) void lra_proj_gemm(
    const float* __restrict__ X, const float* __restrict__ W, int Nout,
    float* __restrict__ outF32, unsigned short* __restrict__ outVh, int mode)
{
  __shared__ unsigned short As[64][40];   // bf16, padded stride
  __shared__ unsigned short Bs[64][40];

  const int tid    = threadIdx.x;
  const int lane   = tid & 31;
  const int wave   = tid >> 5;
  const int half   = lane >> 4;
  const int lane16 = lane & 15;
  const int wm     = wave & 3;            // 4 waves along M
  const int wn     = wave >> 2;           // 2 waves along N
  const int m0     = blockIdx.y * 64;
  const int n0     = blockIdx.x * 64;
  const int lrow   = tid >> 2;            // 64 rows, 4 threads/row
  const int lcol   = (tid & 3) * 8;       // 8 consecutive elements each

  v8f acc0 = {0,0,0,0,0,0,0,0};
  v8f acc1 = {0,0,0,0,0,0,0,0};

  for (int k0 = 0; k0 < D_; k0 += 32) {
    #pragma unroll
    for (int i = 0; i < 8; ++i) {
      As[lrow][lcol + i] = f32_to_bf16(X[(m0 + lrow) * D_ + k0 + lcol + i]);
      Bs[lrow][lcol + i] = f32_to_bf16(W[(n0 + lrow) * D_ + k0 + lcol + i]);
    }
    if (k0 + 32 < D_) {   // hide next slab behind the WMMAs (global_prefetch_b8)
      __builtin_prefetch(&X[(m0 + lrow) * D_ + k0 + 32 + lcol], 0, 0);
      __builtin_prefetch(&W[(n0 + lrow) * D_ + k0 + 32 + lcol], 0, 0);
    }
    __syncthreads();

    BF16x16 a, b0, b1;
    #pragma unroll
    for (int e = 0; e < 16; ++e) {
      a.u[e]  = As[wm * 16 + lane16][kmapA16(e, half)];
      const int kk = kmapA16(e, half);    // B-matrix 32x16 layout, same K-map as A
      b0.u[e] = Bs[wn * 32 + lane16][kk];
      b1.u[e] = Bs[wn * 32 + 16 + lane16][kk];
    }
    acc0 = __builtin_amdgcn_wmma_f32_16x16x32_bf16(false, a.v, false, b0.v,
                                                   (short)0, acc0, false, false);
    acc1 = __builtin_amdgcn_wmma_f32_16x16x32_bf16(false, a.v, false, b1.v,
                                                   (short)0, acc1, false, false);
    __syncthreads();
  }

  #pragma unroll
  for (int r = 0; r < 8; ++r) {
    const int row  = m0 + wm * 16 + r + 8 * half;   // C layout: M = r + 8*half
    const int col0 = n0 + wn * 32 + lane16;
    const int col1 = col0 + 16;
    if (mode == 0) {
      outF32[row * Nout + col0] = acc0[r];
      outF32[row * Nout + col1] = acc1[r];
    } else {
      const int bb = row >> 11, nn = row & (N_ - 1);
      const int h0 = col0 >> 6, d0 = col0 & 63;
      const int h1 = col1 >> 6, d1 = col1 & 63;
      outVh[(((bb * HEADS_ + h0) * N_ + nn) * DH_) + d0] = f32_to_bf16(acc0[r]);
      outVh[(((bb * HEADS_ + h1) * N_ + nn) * DH_) + d1] = f32_to_bf16(acc1[r]);
    }
  }
}

// ---------------------------------------------------------------------------
// Kernel 2: L2-normalize q and k over full rank (256) per row, scatter into
// per-head f32 layout qh/kh:[b][h][n][16]. One wave32 per row.
// ---------------------------------------------------------------------------
__global__ __launch_bounds__(256) void lra_normalize_qk(
    const float* __restrict__ qk, float* __restrict__ qh, float* __restrict__ kh)
{
  const int wave = threadIdx.x >> 5;
  const int lane = threadIdx.x & 31;
  const int row  = blockIdx.x * 8 + wave;     // 0..4095
  const float* rp = qk + (size_t)row * (2 * RANK_);

  float qv[8], kv[8], sq = 0.f, sk = 0.f;
  #pragma unroll
  for (int j = 0; j < 8; ++j) {
    const int e = lane + 32 * j;
    qv[j] = rp[e];          sq += qv[j] * qv[j];
    kv[j] = rp[RANK_ + e];  sk += kv[j] * kv[j];
  }
  #pragma unroll
  for (int off = 16; off > 0; off >>= 1) {
    sq += __shfl_xor(sq, off, 32);
    sk += __shfl_xor(sk, off, 32);
  }
  const float dq = 1.0f / fmaxf(sqrtf(sq), 1e-6f);
  const float dk = 1.0f / fmaxf(sqrtf(sk), 1e-6f);
  const int bb = row >> 11, nn = row & (N_ - 1);
  #pragma unroll
  for (int j = 0; j < 8; ++j) {
    const int e = lane + 32 * j;
    const int idx = ((bb * HEADS_ + (e >> 4)) * N_ + nn) * HS_ + (e & 15);
    qh[idx] = qv[j] * dq;
    kh[idx] = kv[j] * dk;
  }
}

// ---------------------------------------------------------------------------
// Kernel 3: flash-style causal attention, block-cooperative K/V staging.
// Block = 8 waves x 16 q-rows = 128 q rows of one (b,h). All waves share a
// uniform key loop; K (f32) and V (bf16, transposed) tiles are staged to LDS
// once per block with coalesced loads. Tiles past a wave's causal horizon
// self-mask to p=0 in the online-softmax recurrence (step 0 always valid).
// QK^T: 8x V_WMMA_F32_16X16X4_F32 (f32 logits). PV: 4x bf16 WMMA (K=32).
// ---------------------------------------------------------------------------
__global__ __launch_bounds__(256) void lra_attention(
    const float* __restrict__ qh, const float* __restrict__ kh,
    const unsigned short* __restrict__ vh, float* __restrict__ out)
{
  __shared__ float Kt[32][17];                               // 32 keys x 16, pad 17
  __shared__ __align__(16) unsigned short Vt[64][48];        // transposed [dh][key]
  __shared__ __align__(16) unsigned short Pt[8][16][48];     // per-wave P tile

  const int tid    = threadIdx.x;
  const int lane   = tid & 31;
  const int wave   = tid >> 5;
  const int half   = lane >> 4;
  const int lane16 = lane & 15;
  const int bb = blockIdx.y >> 4;
  const int h  = blockIdx.y & 15;
  const int q0 = blockIdx.x * 128 + wave * 16;

  const float*          qbase = qh + (size_t)((bb * HEADS_ + h) * N_) * HS_;
  const float*          kbase = kh + (size_t)((bb * HEADS_ + h) * N_) * HS_;
  const unsigned short* vbase = vh + (size_t)((bb * HEADS_ + h) * N_) * DH_;

  // Q fragment for f32 16x16x4 WMMA: A(16x4): M=lane16, K = r + 2*half (+4s)
  float qf[8];
  {
    const float* qrow = qbase + (q0 + lane16) * HS_;
    #pragma unroll
    for (int s = 0; s < 4; ++s) {
      qf[2 * s]     = qrow[s * 4 + 2 * half];
      qf[2 * s + 1] = qrow[s * 4 + 2 * half + 1];
    }
  }

  v8f acc[4];
  #pragma unroll
  for (int dt = 0; dt < 4; ++dt) acc[dt] = (v8f){0,0,0,0,0,0,0,0};
  float m[8], l[8];
  #pragma unroll
  for (int r = 0; r < 8; ++r) { m[r] = -3.0e38f; l[r] = 0.0f; }

  // staging assignments (uniform across block)
  const int vrow  = tid >> 3;          // 0..31 : key within tile
  const int vcol0 = (tid & 7) * 8;     // 0..56 : dh start, 16B-aligned chunk
  const int ke    = tid * 2;           // K-tile linear element (2 per thread)

  const int jend = blockIdx.x * 128 + 127;   // block causal horizon
  for (int j0 = 0; j0 <= jend; j0 += 32) {
    // ---- cooperative staging: K tile (f32, coalesced linear) ----
    {
      const float* kg = kbase + (size_t)j0 * HS_;
      Kt[ke >> 4][ke & 15]             = kg[ke];
      Kt[(ke + 1) >> 4][(ke + 1) & 15] = kg[ke + 1];
      // V tile: 16B coalesced read, store transposed [dh][key]
      const uint4 vv = *(const uint4*)(vbase + (size_t)(j0 + vrow) * DH_ + vcol0);
      const unsigned int w4[4] = {vv.x, vv.y, vv.z, vv.w};
      #pragma unroll
      for (int i = 0; i < 4; ++i) {
        Vt[vcol0 + 2 * i][vrow]     = (unsigned short)(w4[i] & 0xffffu);
        Vt[vcol0 + 2 * i + 1][vrow] = (unsigned short)(w4[i] >> 16);
      }
    }
    __syncthreads();

    // ---- logits: two 16x16 tiles via f32 WMMA, K=16 in 4 steps ----
    v8f c[2];
    #pragma unroll
    for (int t = 0; t < 2; ++t) {
      v8f ct = {0,0,0,0,0,0,0,0};
      #pragma unroll
      for (int s = 0; s < 4; ++s) {
        v2f a2, b2;
        a2.x = qf[2 * s];  a2.y = qf[2 * s + 1];
        b2.x = Kt[t * 16 + lane16][s * 4 + 2 * half];
        b2.y = Kt[t * 16 + lane16][s * 4 + 2 * half + 1];
        ct = __builtin_amdgcn_wmma_f32_16x16x4_f32(false, a2, false, b2,
                                                   (short)0, ct, false, false);
      }
      #pragma unroll
      for (int r = 0; r < 8; ++r) {        // scale, clip, causal mask
        const int rowq = q0 + r + 8 * half;
        const int key  = j0 + t * 16 + lane16;
        float v = ct[r] * SCALE_;
        v = fminf(fmaxf(v, -10000.0f), 10000.0f);
        if (key > rowq) v = -1.0e30f;
        ct[r] = v;
      }
      c[t] = ct;
    }

    // ---- online softmax over the 32 new keys ----
    float alpha[8];
    #pragma unroll
    for (int r = 0; r < 8; ++r) {
      float tmax = fmaxf(c[0][r], c[1][r]);
      #pragma unroll
      for (int off = 8; off > 0; off >>= 1)
        tmax = fmaxf(tmax, __shfl_xor(tmax, off, 32));
      const float mnew = fmaxf(m[r], tmax);
      const float al   = __expf(m[r] - mnew);
      const float p0   = __expf(c[0][r] - mnew);
      const float p1   = __expf(c[1][r] - mnew);
      float rs = p0 + p1;
      #pragma unroll
      for (int off = 8; off > 0; off >>= 1)
        rs += __shfl_xor(rs, off, 32);
      l[r] = l[r] * al + rs;
      m[r] = mnew;
      alpha[r] = al;
      Pt[wave][r + 8 * half][lane16]      = f32_to_bf16(p0);
      Pt[wave][r + 8 * half][16 + lane16] = f32_to_bf16(p1);
    }
    #pragma unroll
    for (int dt = 0; dt < 4; ++dt)
      #pragma unroll
      for (int r = 0; r < 8; ++r)
        acc[dt][r] *= alpha[r];

    asm volatile("s_wait_dscnt 0" ::: "memory");   // wave-private Pt visible

    BF16x16 pa;   // A fragment 16x32 bf16; K = e + 16*half (contiguous, matches vb)
    #pragma unroll
    for (int e = 0; e < 16; ++e)
      pa.u[e] = Pt[wave][lane16][e + 16 * half];

    #pragma unroll
    for (int dt = 0; dt < 4; ++dt) {
      BF16x16 vb;   // B fragment 32x16: contiguous 32B from transposed Vt
      #pragma unroll
      for (int e = 0; e < 16; ++e)
        vb.u[e] = Vt[dt * 16 + lane16][e + 16 * half];
      acc[dt] = __builtin_amdgcn_wmma_f32_16x16x32_bf16(false, pa.v, false, vb.v,
                                                        (short)0, acc[dt], false, false);
    }
    __syncthreads();   // protect Kt/Vt before next staging round
  }

  // epilogue: divide by row sum (matches reference's renorm; l >= 1 on diag)
  #pragma unroll
  for (int r = 0; r < 8; ++r) {
    const float inv = 1.0f / fmaxf(l[r], 1e-6f);
    const int rowq = q0 + r + 8 * half;
    float* orow = out + (size_t)(bb * N_ + rowq) * D_ + h * DH_;
    #pragma unroll
    for (int dt = 0; dt < 4; ++dt)
      orow[dt * 16 + lane16] = acc[dt][r] * inv;
  }
}

// ---------------------------------------------------------------------------
extern "C" void kernel_launch(void* const* d_in, const int* in_sizes, int n_in,
                              void* d_out, int out_size, void* d_ws, size_t ws_size,
                              hipStream_t stream) {
  const float* x    = (const float*)d_in[0];
  // d_in[1] = mask: unused, causality computed inline (saves 16 MB of reads)
  const float* Wqk  = (const float*)d_in[2];
  const float* Wv   = (const float*)d_in[3];
  float* out        = (float*)d_out;

  char* ws = (char*)d_ws;
  float*          qk_f32 = (float*)(ws);                                     // 8 MB
  float*          qhb    = (float*)(ws + (size_t)8  * 1024 * 1024);          // 4 MB
  float*          khb    = (float*)(ws + (size_t)12 * 1024 * 1024);          // 4 MB
  unsigned short* vhb    = (unsigned short*)(ws + (size_t)16 * 1024 * 1024); // 8 MB

  // qk = x @ Wqk^T  -> f32 [4096, 512]
  lra_proj_gemm<<<dim3((2 * RANK_) / 64, MTOT / 64), 256, 0, stream>>>(
      x, Wqk, 2 * RANK_, qk_f32, nullptr, 0);
  // v  = x @ Wv^T   -> bf16 [b][h][n][64]
  lra_proj_gemm<<<dim3(D_ / 64, MTOT / 64), 256, 0, stream>>>(
      x, Wv, D_, nullptr, vhb, 1);
  // normalize over full rank, scatter to head layout
  lra_normalize_qk<<<MTOT / 8, 256, 0, stream>>>(qk_f32, qhb, khb);
  // flash-style causal attention (block-cooperative K/V staging)
  lra_attention<<<dim3(N_ / 128, B_ * HEADS_), 256, 0, stream>>>(qhb, khb, vhb, out);
}
